// Encoder_5145370821232
// MI455X (gfx1250) — matrix operand order repaired and checked by reading, not compiled, not run on hardware
//
#include <hip/hip_runtime.h>

#define NNODE 50000
#define NEDGE 800000
#define CCH   64
#define EDIM  16
#define ZREAL 144
#define NGR   64

typedef __attribute__((ext_vector_type(16))) __bf16 v16bf;
typedef __attribute__((ext_vector_type(8)))  float  v8f;

// ---- Pre-swizzle W (C x Z row-major, fp32) into bf16 B-matrix fragments ----
// Layout index: ((t*5 + c)*32 + lane)*8 + v ; B 32x16 bf16: lane N=L%16, h=L/16,
// VGPR v holds K = 32c + 8h + 2(v&3) + 16(v>>2) (pair), zero-padded past K=144.
__global__ void pack_w_kernel(const float* __restrict__ W, unsigned* __restrict__ tab) {
    int idx = blockIdx.x * blockDim.x + threadIdx.x;
    if (idx >= 4 * 5 * 32 * 8) return;
    int v    = idx & 7;
    int lane = (idx >> 3) & 31;
    int rest = idx >> 8;
    int c    = rest % 5;
    int t    = rest / 5;
    int h    = lane >> 4;
    int n    = 16 * t + (lane & 15);
    int K    = 32 * c + 8 * h + 2 * (v & 3) + 16 * (v >> 2);
    float f0 = (K     < ZREAL) ? W[n * ZREAL + K]     : 0.f;
    float f1 = (K + 1 < ZREAL) ? W[n * ZREAL + K + 1] : 0.f;
    union { __bf16 b[2]; unsigned u; } p;
    p.b[0] = (__bf16)f0;
    p.b[1] = (__bf16)f1;
    tab[idx] = p.u;
}

// ---- Fused edge GEMM (bf16 WMMA) + gated activation + scatter-add ----
__global__ __launch_bounds__(256)
void edge_kernel(const float* __restrict__ x, const int* __restrict__ ei,
                 const float* __restrict__ ea,
                 const unsigned* __restrict__ tabF, const unsigned* __restrict__ tabS,
                 const float* __restrict__ biasF, const float* __restrict__ biasS,
                 float* __restrict__ acc)
{
    const int lane  = threadIdx.x & 31;
    const int wave  = blockIdx.x * (blockDim.x >> 5) + (threadIdx.x >> 5);
    const long ebase = (long)wave * 16;
    const int Mlane = lane & 15;      // edge row this lane gathers for
    const int h     = lane >> 4;
    const long e    = ebase + Mlane;

    const int srcn = ei[e];            // edge_index[0][e]
    const int dstn = ei[NEDGE + e];    // edge_index[1][e]
    const float* xd  = x  + (size_t)dstn * CCH;
    const float* xs  = x  + (size_t)srcn * CCH;
    const float* eap = ea + (size_t)e * EDIM;

    v8f accf[4] = {};
    v8f accs[4] = {};

    #pragma unroll
    for (int c = 0; c < 5; ++c) {
        // Each lane needs two contiguous 8-float runs of z[e][.]:
        //   group0: K = 32c + 8h .. +7   (A VGPRs 0-3)
        //   group1: K = 32c + 16 + 8h .. +7 (A VGPRs 4-7)
        float g0f[8], g1f[8];
        if (c < 4) {
            const float* p = (c < 2) ? xd : xs;
            const int off0 = 32 * c + 8 * h - ((c < 2) ? 0 : 64);
            float4 a0 = ((const float4*)(p + off0))[0];
            float4 a1 = ((const float4*)(p + off0))[1];
            float4 b0 = ((const float4*)(p + off0 + 16))[0];
            float4 b1 = ((const float4*)(p + off0 + 16))[1];
            g0f[0]=a0.x; g0f[1]=a0.y; g0f[2]=a0.z; g0f[3]=a0.w;
            g0f[4]=a1.x; g0f[5]=a1.y; g0f[6]=a1.z; g0f[7]=a1.w;
            g1f[0]=b0.x; g1f[1]=b0.y; g1f[2]=b0.z; g1f[3]=b0.w;
            g1f[4]=b1.x; g1f[5]=b1.y; g1f[6]=b1.z; g1f[7]=b1.w;
        } else {
            // K in [128,144): edge_attr; K >= 144: zero pad
            float4 a0 = ((const float4*)(eap + 8 * h))[0];
            float4 a1 = ((const float4*)(eap + 8 * h))[1];
            g0f[0]=a0.x; g0f[1]=a0.y; g0f[2]=a0.z; g0f[3]=a0.w;
            g0f[4]=a1.x; g0f[5]=a1.y; g0f[6]=a1.z; g0f[7]=a1.w;
            #pragma unroll
            for (int i = 0; i < 8; ++i) g1f[i] = 0.f;
        }
        v16bf Av;
        #pragma unroll
        for (int i = 0; i < 8; ++i) {
            Av[i]     = (__bf16)g0f[i];   // native v_cvt bf16 path
            Av[8 + i] = (__bf16)g1f[i];
        }
        #pragma unroll
        for (int t = 0; t < 4; ++t) {
            const v16bf bF = *(const v16bf*)(tabF + (((t * 5 + c) * 32 + lane) * 8));
            const v16bf bS = *(const v16bf*)(tabS + (((t * 5 + c) * 32 + lane) * 8));
            accf[t] = __builtin_amdgcn_wmma_f32_16x16x32_bf16(false, Av, false, bF,
                                                              (short)0, accf[t], false, false);
            accs[t] = __builtin_amdgcn_wmma_f32_16x16x32_bf16(false, Av, false, bS,
                                                              (short)0, accs[t], false, false);
        }
    }

    // D layout: lane L holds (M = v + 8h, N = L%16). Gather dst per row once (int4 x2).
    int dstrow[8];
    {
        const int4* dp = (const int4*)(ei + NEDGE + ebase + 8 * h);
        int4 d0 = dp[0], d1 = dp[1];
        dstrow[0]=d0.x; dstrow[1]=d0.y; dstrow[2]=d0.z; dstrow[3]=d0.w;
        dstrow[4]=d1.x; dstrow[5]=d1.y; dstrow[6]=d1.z; dstrow[7]=d1.w;
    }

    #pragma unroll
    for (int t = 0; t < 4; ++t) {
        const int ch = 16 * t + Mlane;
        const float bF = biasF[ch], bS = biasS[ch];
        #pragma unroll
        for (int v = 0; v < 8; ++v) {
            float f  = accf[t][v] + bF;
            float s  = accs[t][v] + bS;
            float sg = 1.f / (1.f + __expf(-f));
            float sp = (s > 15.f) ? s : __logf(1.f + __expf(s));
            atomicAdd(&acc[(size_t)dstrow[v] * CCH + ch], sg * sp);
        }
    }
}

// ---- helpers ----
__global__ void copy_kernel(const float* __restrict__ in, float* __restrict__ out, long n) {
    long i = (long)blockIdx.x * blockDim.x + threadIdx.x;
    if (i < n) out[i] = in[i];
}

__global__ void bn_stats_kernel(const float* __restrict__ xb, float* __restrict__ stats) {
    int ch   = threadIdx.x & 63;
    int row0 = blockIdx.x * (blockDim.x >> 6) + (threadIdx.x >> 6);
    int strd = gridDim.x * (blockDim.x >> 6);
    float s = 0.f, ss = 0.f;
    for (int r = row0; r < NNODE; r += strd) {
        float v = xb[(size_t)r * CCH + ch];
        s += v; ss += v * v;
    }
    atomicAdd(&stats[ch], s);
    atomicAdd(&stats[CCH + ch], ss);
}

__global__ void bn_apply_kernel(const float* __restrict__ xin, float* __restrict__ xout,
                                const float* __restrict__ stats,
                                const float* __restrict__ gamma, const float* __restrict__ beta) {
    long i = (long)blockIdx.x * blockDim.x + threadIdx.x;
    if (i >= (long)NNODE * CCH) return;
    int ch   = (int)(i & 63);
    float m  = stats[ch] * (1.f / NNODE);
    float vr = stats[CCH + ch] * (1.f / NNODE) - m * m;
    float iv = rsqrtf(vr + 1e-5f);
    xout[i] = (xin[i] - m) * iv * gamma[ch] + beta[ch];
}

__global__ void pool_acc_kernel(const float* __restrict__ xb, const int* __restrict__ batch,
                                float* __restrict__ sums, float* __restrict__ cnt) {
    long i = (long)blockIdx.x * blockDim.x + threadIdx.x;
    if (i >= (long)NNODE * CCH) return;
    int n = (int)(i >> 6), ch = (int)(i & 63);
    int g = batch[n];
    atomicAdd(&sums[(size_t)g * CCH + ch], xb[i]);
    if (ch == 0) atomicAdd(&cnt[g], 1.f);
}

__global__ void pool_div_kernel(const float* __restrict__ sums, const float* __restrict__ cnt,
                                float* __restrict__ out) {
    int i = blockIdx.x * blockDim.x + threadIdx.x;
    if (i < NGR * CCH) out[i] = sums[i] / fmaxf(cnt[i >> 6], 1.f);
}

extern "C" void kernel_launch(void* const* d_in, const int* in_sizes, int n_in,
                              void* d_out, int out_size, void* d_ws, size_t ws_size,
                              hipStream_t stream) {
    const float* x     = (const float*)d_in[0];
    const int*   ei    = (const int*)d_in[1];
    const float* ea    = (const float*)d_in[2];
    const int*   batch = (const int*)d_in[3];
    const float *Wf0 = (const float*)d_in[4],  *bf0 = (const float*)d_in[5];
    const float *Ws0 = (const float*)d_in[6],  *bs0 = (const float*)d_in[7];
    const float *g0  = (const float*)d_in[8],  *be0 = (const float*)d_in[9];
    const float *Wf1 = (const float*)d_in[10], *bf1 = (const float*)d_in[11];
    const float *Ws1 = (const float*)d_in[12], *bs1 = (const float*)d_in[13];
    const float *g1  = (const float*)d_in[14], *be1 = (const float*)d_in[15];

    char* ws = (char*)d_ws;
    const size_t NC_BYTES = (size_t)NNODE * CCH * sizeof(float);   // 12.8 MB
    float*    bufA  = (float*)ws;
    float*    bufB  = (float*)(ws + NC_BYTES);
    float*    stats = (float*)(ws + 2 * NC_BYTES);                 // 128 floats
    float*    sums  = (float*)(ws + 2 * NC_BYTES + 512);           // 4096 floats
    float*    cnt   = (float*)(ws + 2 * NC_BYTES + 512 + 16384);   // 64 floats
    unsigned* tab   = (unsigned*)(ws + 2 * NC_BYTES + 512 + 16384 + 256);
    unsigned *tF0 = tab, *tS0 = tab + 5120, *tF1 = tab + 10240, *tS1 = tab + 15360;

    const long NC = (long)NNODE * CCH;                // 3.2M
    const int  CPB = 256;
    const int  NC_BLOCKS = (int)((NC + CPB - 1) / CPB);

    // Pack weights into bf16 B-fragment tables
    pack_w_kernel<<<20, 256, 0, stream>>>(Wf0, tF0);
    pack_w_kernel<<<20, 256, 0, stream>>>(Ws0, tS0);
    pack_w_kernel<<<20, 256, 0, stream>>>(Wf1, tF1);
    pack_w_kernel<<<20, 256, 0, stream>>>(Ws1, tS1);

    // ---- Layer 0 ----
    copy_kernel<<<NC_BLOCKS, CPB, 0, stream>>>(x, bufA, NC);       // residual base
    hipMemsetAsync(stats, 0, 512, stream);
    edge_kernel<<<NEDGE / 128, 256, 0, stream>>>(x, ei, ea, tF0, tS0, bf0, bs0, bufA);
    bn_stats_kernel<<<512, 256, 0, stream>>>(bufA, stats);
    bn_apply_kernel<<<NC_BLOCKS, CPB, 0, stream>>>(bufA, bufB, stats, g0, be0);

    // ---- Layer 1 ----
    copy_kernel<<<NC_BLOCKS, CPB, 0, stream>>>(bufB, bufA, NC);    // residual base (avoid RW race)
    hipMemsetAsync(stats, 0, 512, stream);
    edge_kernel<<<NEDGE / 128, 256, 0, stream>>>(bufB, ei, ea, tF1, tS1, bf1, bs1, bufA);
    bn_stats_kernel<<<512, 256, 0, stream>>>(bufA, stats);
    bn_apply_kernel<<<NC_BLOCKS, CPB, 0, stream>>>(bufA, bufB, stats, g1, be1);

    // ---- Global mean pool ----
    hipMemsetAsync(sums, 0, 16384 + 256, stream);
    pool_acc_kernel<<<NC_BLOCKS, CPB, 0, stream>>>(bufB, batch, sums, cnt);
    pool_div_kernel<<<16, 256, 0, stream>>>(sums, cnt, (float*)d_out);
}
